// RNN_85529978732704
// MI455X (gfx1250) — matrix operand order repaired and checked by reading, not compiled, not run on hardware
//
#include <hip/hip_runtime.h>
#include <math.h>

// ---------------------------------------------------------------------------
// Fused tanh-RNN for MI455X (gfx1250, wave32, WMMA bf16 16x16x32)
//   B=128, T=4096, I=64, H=256, O=64
//   grid = 8 WGs (16 batches each), 256 threads (8 waves) per WG
//   - W_hh (128 VGPRs/lane) and W_ih (32 VGPRs/lane) register-resident as
//     bf16 WMMA B-fragments (uses gfx1250 extended VGPR file via VGPR-MSB)
//   - hidden state ping-pongs through padded row-major LDS
//   - split workgroup barrier (s_barrier_signal/-wait) with the x f32->bf16
//     conversion overlapped inside the barrier window
//   - input-projection WMMAs issued first each step to cover ds_load latency
// ---------------------------------------------------------------------------

typedef __attribute__((ext_vector_type(16))) __bf16 v16bf;
typedef __attribute__((ext_vector_type(8)))  __bf16 v8bf;
typedef __attribute__((ext_vector_type(8)))  float  v8f;
typedef __attribute__((ext_vector_type(4)))  float  f4;

#define WMMA_BF16(a, b, c)                                                   \
  __builtin_amdgcn_wmma_f32_16x16x32_bf16(false, (a), false, (b), (short)0,  \
                                          (c), false, false)

#if __has_builtin(__builtin_amdgcn_s_barrier_signal) && \
    __has_builtin(__builtin_amdgcn_s_barrier_wait)
#define RNN_SPLIT_BARRIER 1
#else
#define RNN_SPLIT_BARRIER 0
#endif

__device__ __forceinline__ float fast_tanh(float x) {
#if __has_builtin(__builtin_amdgcn_tanhf)
  return __builtin_amdgcn_tanhf(x);
#elif __has_builtin(__builtin_amdgcn_tanh_f32)
  return __builtin_amdgcn_tanh_f32(x);
#else
  return tanhf(x);
#endif
}

namespace {
constexpr int kT = 4096;
constexpr int kI = 64;
constexpr int kH = 256;
constexpr int kO = 64;
constexpr int kPR = kH + 8;  // padded LDS row (bf16): 132 dwords -> clean banks
constexpr int kPF = 8;       // L2/L0 prefetch distance (timesteps)
}  // namespace

// Load a [16-col, K=32] bf16 WMMA B-fragment from a row-major f32 matrix row.
__device__ __forceinline__ v16bf load_b_frag(const float* wrow, int k0) {
  f4 w0 = *(const f4*)(wrow + k0);
  f4 w1 = *(const f4*)(wrow + k0 + 4);
  f4 w2 = *(const f4*)(wrow + k0 + 16);
  f4 w3 = *(const f4*)(wrow + k0 + 20);
  v16bf b;
#pragma unroll
  for (int j = 0; j < 4; ++j) {
    b[j]      = (__bf16)w0[j];
    b[4 + j]  = (__bf16)w1[j];
    b[8 + j]  = (__bf16)w2[j];
    b[12 + j] = (__bf16)w3[j];
  }
  return b;
}

__global__ __launch_bounds__(256, 1) void rnn_fused_kernel(
    const float* __restrict__ x,      // [B, T, I]
    const float* __restrict__ Wih,    // [H, I]
    const float* __restrict__ Whh,    // [H, H]
    const float* __restrict__ bih,    // [H]
    const float* __restrict__ bhh,    // [H]
    const float* __restrict__ Wy,     // [O, H]
    const float* __restrict__ by,     // [O]
    float* __restrict__ out)          // [B, O]
{
  __shared__ alignas(16) __bf16 hbuf[2][16][kPR];

  const int tid   = threadIdx.x;
  const int wave  = tid >> 5;
  const int lane  = tid & 31;
  const int lm    = lane & 15;   // M row (A) or N col (B/C) selector
  const int lh    = lane >> 4;   // half-wave selector
  const int btile = blockIdx.x;  // 0..7

  // ---- W_hh -> register-resident bf16 B fragments: Bhh[kt][nt]
  v16bf Bhh[8][2];
#pragma unroll
  for (int nt = 0; nt < 2; ++nt) {
    const float* wrow = Whh + (size_t)(wave * 32 + nt * 16 + lm) * kH;
#pragma unroll
    for (int kt = 0; kt < 8; ++kt)
      Bhh[kt][nt] = load_b_frag(wrow, kt * 32 + lh * 8);
  }

  // ---- W_ih -> register-resident bf16 B fragments (K=64 -> 2 k-ops)
  v16bf Bih[2][2];
#pragma unroll
  for (int nt = 0; nt < 2; ++nt) {
    const float* wrow = Wih + (size_t)(wave * 32 + nt * 16 + lm) * kI;
#pragma unroll
    for (int kt = 0; kt < 2; ++kt)
      Bih[kt][nt] = load_b_frag(wrow, kt * 32 + lh * 8);
  }

  // ---- bias splat vectors (C-layout: all 8 values of a lane share N)
  v8f biasv0, biasv1;
  {
    const int n0 = wave * 32 + lm;
    const float b0 = bih[n0] + bhh[n0];
    const float b1 = bih[n0 + 16] + bhh[n0 + 16];
#pragma unroll
    for (int j = 0; j < 8; ++j) {
      biasv0[j] = b0;
      biasv1[j] = b1;
    }
  }

  // ---- h(-1) = 0
  for (int i = tid; i < 16 * kPR; i += 256) (&hbuf[0][0][0])[i] = (__bf16)0.0f;
  __syncthreads();

  // ---- x prefetch state: f32 A-fragment staging + packed bf16 A operands
  const float* xlane = x + (size_t)(btile * 16 + lm) * kT * kI;
  const int c0 = lh * 8;  // A layout: lanes 0-15 -> K {c..c+7, c+16..c+23}
  f4 xf[8];
  v16bf ax0, ax1;
  {
    const float* xp = xlane;  // t = 0
    xf[0] = *(const f4*)(xp + c0);            xf[1] = *(const f4*)(xp + c0 + 4);
    xf[2] = *(const f4*)(xp + c0 + 16);       xf[3] = *(const f4*)(xp + c0 + 20);
    xf[4] = *(const f4*)(xp + 32 + c0);       xf[5] = *(const f4*)(xp + 32 + c0 + 4);
    xf[6] = *(const f4*)(xp + 32 + c0 + 16);  xf[7] = *(const f4*)(xp + 32 + c0 + 20);
#pragma unroll
    for (int j = 0; j < 4; ++j) {
      ax0[j]      = (__bf16)xf[0][j];
      ax0[4 + j]  = (__bf16)xf[1][j];
      ax0[8 + j]  = (__bf16)xf[2][j];
      ax0[12 + j] = (__bf16)xf[3][j];
      ax1[j]      = (__bf16)xf[4][j];
      ax1[4 + j]  = (__bf16)xf[5][j];
      ax1[8 + j]  = (__bf16)xf[6][j];
      ax1[12 + j] = (__bf16)xf[7][j];
    }
  }

  // =========================== recurrence over T ===========================
  for (int t = 0; t < kT; ++t) {
    // ---- input projection FIRST: register-only operands, no LDS dependency,
    //      covers the latency of the A-fragment ds_loads issued below.
    v8f acc0 = WMMA_BF16(ax0, Bih[0][0], biasv0);
    v8f acc1 = WMMA_BF16(ax0, Bih[0][1], biasv1);
    acc0 = WMMA_BF16(ax1, Bih[1][0], acc0);
    acc1 = WMMA_BF16(ax1, Bih[1][1], acc1);

    // ---- issue next step's x loads (consumed at end of this iteration)
    if (t + 1 < kT) {
      const float* xp = xlane + (size_t)(t + 1) * kI;
      xf[0] = *(const f4*)(xp + c0);            xf[1] = *(const f4*)(xp + c0 + 4);
      xf[2] = *(const f4*)(xp + c0 + 16);       xf[3] = *(const f4*)(xp + c0 + 20);
      xf[4] = *(const f4*)(xp + 32 + c0);       xf[5] = *(const f4*)(xp + 32 + c0 + 4);
      xf[6] = *(const f4*)(xp + 32 + c0 + 16);  xf[7] = *(const f4*)(xp + 32 + c0 + 20);
    }
    // ---- deep prefetch into L0/L2 (global_prefetch_b8), no VGPR cost
    if (t + kPF < kT) {
      __builtin_prefetch(xlane + (size_t)(t + kPF) * kI + c0, 0, 1);
    }

    // ---- h @ W_hh^T : A fragments gathered from padded row-major LDS
    const __bf16* hb = &hbuf[t & 1][0][0];
#pragma unroll
    for (int kt = 0; kt < 8; ++kt) {
      const __bf16* p = hb + lm * kPR + kt * 32 + lh * 8;
      v8bf lo = *(const v8bf*)p;
      v8bf hi = *(const v8bf*)(p + 16);
      v16bf a;
#pragma unroll
      for (int j = 0; j < 8; ++j) {
        a[j]     = lo[j];
        a[8 + j] = hi[j];
      }
      acc0 = WMMA_BF16(a, Bhh[kt][0], acc0);
      acc1 = WMMA_BF16(a, Bhh[kt][1], acc1);
    }

    // ---- tanh -> bf16 -> next LDS buffer (row-major; C layout: M=j+8*lh)
    __bf16* hn = &hbuf[(t + 1) & 1][0][0];
    const int n0 = wave * 32 + lm;
    const int mb = lh * 8;
#pragma unroll
    for (int j = 0; j < 8; ++j) {
      hn[(mb + j) * kPR + n0]      = (__bf16)fast_tanh(acc0[j]);
      hn[(mb + j) * kPR + n0 + 16] = (__bf16)fast_tanh(acc1[j]);
    }

    // ---- barrier; overlap next step's x cvt inside the barrier window
#if RNN_SPLIT_BARRIER
    __builtin_amdgcn_fence(__ATOMIC_RELEASE, "workgroup");
    __builtin_amdgcn_s_barrier_signal(-1);
#endif
    if (t + 1 < kT) {
#pragma unroll
      for (int j = 0; j < 4; ++j) {
        ax0[j]      = (__bf16)xf[0][j];
        ax0[4 + j]  = (__bf16)xf[1][j];
        ax0[8 + j]  = (__bf16)xf[2][j];
        ax0[12 + j] = (__bf16)xf[3][j];
        ax1[j]      = (__bf16)xf[4][j];
        ax1[4 + j]  = (__bf16)xf[5][j];
        ax1[8 + j]  = (__bf16)xf[6][j];
        ax1[12 + j] = (__bf16)xf[7][j];
      }
    }
#if RNN_SPLIT_BARRIER
    __builtin_amdgcn_s_barrier_wait(-1);
    __builtin_amdgcn_fence(__ATOMIC_ACQUIRE, "workgroup");
#else
    __syncthreads();
#endif
  }

  // ====================== output: y = h_T @ W_y^T + b_y ====================
  // T even -> h_T sits in hbuf[0]. O=64 -> 4 N-tiles, waves 0..3.
  if (wave < 4) {
    const int o = wave * 16 + lm;
    const float* wrow = Wy + (size_t)o * kH;
    v8f acc;
    const float b = by[o];
#pragma unroll
    for (int j = 0; j < 8; ++j) acc[j] = b;

    const __bf16* hb = &hbuf[0][0][0];
#pragma unroll
    for (int kt = 0; kt < 8; ++kt) {
      v16bf bb = load_b_frag(wrow, kt * 32 + lh * 8);
      const __bf16* p = hb + lm * kPR + kt * 32 + lh * 8;
      v8bf lo = *(const v8bf*)p;
      v8bf hi = *(const v8bf*)(p + 16);
      v16bf a;
#pragma unroll
      for (int j = 0; j < 8; ++j) {
        a[j]     = lo[j];
        a[8 + j] = hi[j];
      }
      acc = WMMA_BF16(a, bb, acc);
    }
#pragma unroll
    for (int j = 0; j < 8; ++j) {
      const int bi = btile * 16 + lh * 8 + j;
      out[bi * kO + o] = acc[j];
    }
  }
}

// ---------------------------------------------------------------------------
extern "C" void kernel_launch(void* const* d_in, const int* in_sizes, int n_in,
                              void* d_out, int out_size, void* d_ws,
                              size_t ws_size, hipStream_t stream) {
  (void)in_sizes; (void)n_in; (void)out_size; (void)d_ws; (void)ws_size;
  const float* x   = (const float*)d_in[0];
  const float* wih = (const float*)d_in[1];
  const float* whh = (const float*)d_in[2];
  const float* bih = (const float*)d_in[3];
  const float* bhh = (const float*)d_in[4];
  const float* wy  = (const float*)d_in[5];
  const float* by  = (const float*)d_in[6];
  float* out = (float*)d_out;

  rnn_fused_kernel<<<dim3(8), dim3(256), 0, stream>>>(x, wih, whh, bih, bhh,
                                                      wy, by, out);
}
